// SmallSpatialPolicyHead_40269613367302
// MI455X (gfx1250) — compile-verified, matrix operand order checked by reading, not run on hardware
//
#include <hip/hip_runtime.h>
#include <hip/hip_bf16.h>
#include <math.h>

typedef __bf16 bf16_t;
typedef __attribute__((ext_vector_type(16))) __bf16 v16bf;
typedef __attribute__((ext_vector_type(8)))  __bf16 v8bf;
typedef __attribute__((ext_vector_type(4)))  __bf16 v4bf;
typedef __attribute__((ext_vector_type(8)))  float  v8f;

union Frag16 { v16bf v; v8bf h[2]; };

__device__ __forceinline__ v8f wmma_bf16(v16bf a, v16bf b, v8f c) {
  // 8 args: (neg_a, A, neg_b, B, c_mod, C, reuse_a, reuse_b)
  return __builtin_amdgcn_wmma_f32_16x16x32_bf16(false, a, false, b, (short)0, c, false, false);
}

__device__ __forceinline__ v8f vzero8() { v8f z = {}; return z; }

__device__ __forceinline__ float fast_tanh(float x) {
#if __has_builtin(__builtin_amdgcn_tanhf)
  return __builtin_amdgcn_tanhf(x);
#elif __has_builtin(__builtin_amdgcn_tanh_f32)
  return __builtin_amdgcn_tanh_f32(x);
#else
  return tanhf(x);
#endif
}

__device__ __forceinline__ float mish_f(float x) {
  float sp = (x > 15.f) ? x : __logf(1.f + __expf(x));
  return x * fast_tanh(sp);
}

// Async 16B global->LDS copy (GV mode). LDS byte offset = low 32 bits of the
// generic pointer (flat LDS aperture maps addr[31:0] to the LDS address).
__device__ __forceinline__ void async_cp16(void* lds, const void* g) {
  unsigned l = (unsigned)(uintptr_t)lds;
  asm volatile("global_load_async_to_lds_b128 %0, %1, off" :: "v"(l), "v"(g) : "memory");
}
__device__ __forceinline__ void async_wait0() {
  asm volatile("s_wait_asynccnt 0x0" ::: "memory");
}

__device__ __forceinline__ v16bf load_frag(const bf16_t* p0, const bf16_t* p1) {
  Frag16 f;
  f.h[0] = *(const v8bf*)p0;
  f.h[1] = *(const v8bf*)p1;
  return f.v;
}

// Operand fragment from row-major W[rows][Kp]; per-lane 16-bit 16x16x32 pattern:
//   elems [0..7]  = row[kstep*32 + half*8 .. +7]
//   elems [8..15] = row[kstep*32 + half*8 + 16 .. +23]
__device__ __forceinline__ v16bf bfrag(const bf16_t* W, int Kp, int n, int kstep, int half) {
  const bf16_t* p = W + n * Kp + kstep * 32 + half * 8;
  return load_frag(p, p + 16);
}

// ---------------- constants ----------------
#define NNODE  54
#define TDIM   80
#define ADIM   397

// ws layout (bytes)
#define OFF_TN   0u                 // bf16 [B][80]
#define OFF_H    2621440u           // f32  [B][80]
#define OFF_PS   7864320u           // f32  [1024][80]
#define OFF_PQ   8192000u           // f32  [1024][80]
#define OFF_SC   8519680u           // f32  [80]
#define OFF_SHF  8520000u           // f32  [80]
#define OFF_G1   8520320u           // bf16 [80][96]
#define OFF_G2   8535680u           // bf16 [400][96]
#define OFF_W1   8612480u           // bf16 sett1|city1|road1p|rob1 = 18432 elems

// dynamic LDS layout for k_heads (bytes, all 16B aligned)
#define L_TN    0u        // bf16 [16][80]      2560
#define L_HM    2560u     // bf16 [16][96]      3072
#define L_NN    5632u     // bf16 [16][54][48]  82944
#define L_TILE  88576u    // bf16 [16][19][48]  29184
#define L_W1    117760u   // bf16 18432 elems   36864
#define L_ZP    154624u   // bf16 [16]          32
#define LDS_BYTES 154656u

// ---------------- prep: weights -> padded bf16 ----------------
__global__ void k_prep(const float* gfc1_w, const float* gfc2_w,
                       const float* sett1_w, const float* city1_w,
                       const float* road1_w, const float* rob1_w,
                       bf16_t* gfc1_bf, bf16_t* gfc2_bf, bf16_t* w1_all) {
  int tid = threadIdx.x;
  for (int i = tid; i < 80 * 96; i += 256) {
    int r = i / 96, c = i % 96;
    gfc1_bf[i] = (bf16_t)((c < 80) ? gfc1_w[r * 80 + c] : 0.f);
  }
  for (int i = tid; i < 400 * 96; i += 256) {
    int r = i / 96, c = i % 96;
    gfc2_bf[i] = (bf16_t)((r < 397 && c < 80) ? gfc2_w[r * 80 + c] : 0.f);
  }
  for (int i = tid; i < 32 * 128; i += 256) w1_all[i]         = (bf16_t)sett1_w[i];
  for (int i = tid; i < 32 * 128; i += 256) w1_all[4096 + i]  = (bf16_t)city1_w[i];
  for (int i = tid; i < 32 * 192; i += 256) {
    int r = i / 192, c = i % 192;
    w1_all[8192 + i] = (bf16_t)((c < 176) ? road1_w[r * 176 + c] : 0.f);
  }
  for (int i = tid; i < 32 * 128; i += 256) w1_all[14336 + i] = (bf16_t)rob1_w[i];
}

// ---------------- trunk LN + gemm1 + BN partial stats ----------------
__global__ __launch_bounds__(128) void k_trunk(
    const float* trunk, const float* lnw, const float* lnb,
    const bf16_t* gfc1_bf, const float* gfc1_b,
    bf16_t* tn_g, float* hbuf, float* psum, float* psq) {
  __shared__ bf16_t tn_l[16 * 96];
  int tid = threadIdx.x;
  int b0 = blockIdx.x * 16;
  if (tid < 16) {
    int row = tid;
    const float4* x4 = (const float4*)(trunk + (b0 + row) * 80);
    float v[80];
    float s = 0.f, ss = 0.f;
#pragma unroll
    for (int q = 0; q < 20; q++) {
      float4 f = x4[q];
      v[4*q] = f.x; v[4*q+1] = f.y; v[4*q+2] = f.z; v[4*q+3] = f.w;
    }
#pragma unroll
    for (int k = 0; k < 80; k++) { s += v[k]; ss += v[k] * v[k]; }
    float mu = s * (1.f / 80.f);
    float var = ss * (1.f / 80.f) - mu * mu;
    float rs = rsqrtf(var + 1e-5f);
#pragma unroll
    for (int q = 0; q < 10; q++) {
      v8bf o;
#pragma unroll
      for (int e = 0; e < 8; e++) {
        int k = q * 8 + e;
        o[e] = (bf16_t)((v[k] - mu) * rs * lnw[k] + lnb[k]);
      }
      *(v8bf*)(tn_l + row * 96 + q * 8) = o;
      *(v8bf*)(tn_g + (b0 + row) * 80 + q * 8) = o;
    }
    v8bf z8 = {};
    *(v8bf*)(tn_l + row * 96 + 80) = z8;
    *(v8bf*)(tn_l + row * 96 + 88) = z8;
  }
  __syncthreads();
  int lane = tid & 31, wid = tid >> 5;
  int l16 = lane & 15, half = lane >> 4;
  for (int nt = wid; nt < 5; nt += 4) {
    v8f d = vzero8();
#pragma unroll
    for (int t = 0; t < 3; t++) {
      const bf16_t* p = tn_l + l16 * 96 + t * 32 + half * 8;
      v16bf a = load_frag(p, p + 16);
      v16bf b = bfrag(gfc1_bf, 96, nt * 16 + l16, t, half);
      d = wmma_bf16(a, b, d);
    }
    int col = nt * 16 + l16;
    float bias = gfc1_b[col];
    float s = 0.f, ss = 0.f;
#pragma unroll
    for (int i = 0; i < 8; i++) {
      float v = d[i] + bias;
      hbuf[(b0 + i + 8 * half) * 80 + col] = v;
      s += v; ss += v * v;
    }
    s  += __shfl_xor(s, 16, 32);
    ss += __shfl_xor(ss, 16, 32);
    if (half == 0) {
      psum[blockIdx.x * 80 + col] = s;
      psq [blockIdx.x * 80 + col] = ss;
    }
  }
}

// ---------------- finalize BN stats ----------------
__global__ void k_bnstats(const float* psum, const float* psq,
                          const float* bn_w, const float* bn_b,
                          float* scale, float* shift, int nblk, float invB) {
  int f = threadIdx.x;
  if (f >= 80) return;
  float s = 0.f, ss = 0.f;
  for (int b = 0; b < nblk; b++) { s += psum[b * 80 + f]; ss += psq[b * 80 + f]; }
  float mu = s * invB;
  float var = ss * invB - mu * mu;
  float sc = rsqrtf(var + 1e-5f) * bn_w[f];
  scale[f] = sc;
  shift[f] = bn_b[f] - mu * sc;
}

// ---------------- fused heads ----------------
struct CParams {
  float* out;
  const bf16_t* tn_g;
  const float* hbuf;
  const float* scale;
  const float* shift;
  const float* node_emb;
  const float* ln_n_w;
  const float* ln_n_b;
  const int* road_pairs;
  const int* tile_nodes;
  const bf16_t* w1_all_g;
  const bf16_t* gfc2_bf;
  const float* gfc2_b;
  const float* sett1_b; const float* sett2_w; const float* sett2_b;
  const float* city1_b; const float* city2_w; const float* city2_b;
  const float* road1_b; const float* road2_w; const float* road2_b;
  const float* rob1_b;  const float* rob2_w;  const float* rob2_b;
};

// Second layer (SH=32 -> 1): mish + dot(w2) + butterfly reduce over 16 lanes.
__device__ __forceinline__ void head1(const v8f& y0, const v8f& y1,
    float b1c0, float b1c1, float w0, float w1, float b2,
    float* out, int b0, int col, int lane) {
  int l16 = lane & 15, half = lane >> 4;
  float acc[8];
#pragma unroll
  for (int i = 0; i < 8; i++)
    acc[i] = mish_f(y0[i] + b1c0) * w0 + mish_f(y1[i] + b1c1) * w1;
#pragma unroll
  for (int m = 1; m < 16; m <<= 1) {
#pragma unroll
    for (int i = 0; i < 8; i++) acc[i] += __shfl_xor(acc[i], m, 32);
  }
  if (l16 == 0) {
#pragma unroll
    for (int i = 0; i < 8; i++)
      out[(size_t)(b0 + i + 8 * half) * ADIM + col] = acc[i] + b2;
  }
}

__device__ __forceinline__ v16bf ctx_frag_node(const bf16_t* tn_l, const bf16_t* nn_l,
                                               int node, int l16, int kstep, int half) {
  int s0 = kstep * 32 + half * 8;
  int s1 = s0 + 16;
  const bf16_t* bt = tn_l + l16 * 80;
  const bf16_t* bn = nn_l + (l16 * 54 + node) * 48 - 80;   // virtual ctx offset
  const bf16_t* p0 = (s0 < 80) ? bt + s0 : bn + s0;
  const bf16_t* p1 = (s1 < 80) ? bt + s1 : bn + s1;
  return load_frag(p0, p1);
}

__device__ __forceinline__ const bf16_t* road_ptr(const bf16_t* tn_l, const bf16_t* nn_l,
    const bf16_t* zp, int sn, int dn, int l16, int s) {
  if (s >= 176) return zp;
  if (s < 80)   return tn_l + l16 * 80 + s;
  if (s < 128)  return nn_l + (l16 * 54 + sn) * 48 + (s - 80);
  return nn_l + (l16 * 54 + dn) * 48 + (s - 128);
}

__global__ __launch_bounds__(256) void k_heads(CParams P) {
  extern __shared__ char smem[];
  bf16_t* tn_l   = (bf16_t*)(smem + L_TN);
  bf16_t* hm_l   = (bf16_t*)(smem + L_HM);
  bf16_t* nn_l   = (bf16_t*)(smem + L_NN);
  bf16_t* tile_l = (bf16_t*)(smem + L_TILE);
  bf16_t* w1_l   = (bf16_t*)(smem + L_W1);
  bf16_t* zp     = (bf16_t*)(smem + L_ZP);

  int tid = threadIdx.x;
  int b0 = blockIdx.x * 16;

  // --- async DMA staging: tn tile (160 x 16B) + layer-1 weights (2304 x 16B) ---
  {
    const bf16_t* tsrc = P.tn_g + b0 * 80;
    for (int i = tid; i < 160; i += 256)
      async_cp16(tn_l + i * 8, tsrc + i * 8);
    for (int i = tid; i < 2304; i += 256)
      async_cp16(w1_l + i * 8, P.w1_all_g + i * 8);
  }
  if (tid < 16) zp[tid] = (bf16_t)0.f;

  // mish(batchnorm(h)) -> bf16, padded to 96
  for (int i = tid; i < 16 * 20; i += 256) {
    int row = i / 20, q = i % 20;
    float4 h4 = ((const float4*)(P.hbuf + (b0 + row) * 80))[q];
    float4 sc = ((const float4*)P.scale)[q];
    float4 sh = ((const float4*)P.shift)[q];
    v4bf o;
    o[0] = (bf16_t)mish_f(h4.x * sc.x + sh.x);
    o[1] = (bf16_t)mish_f(h4.y * sc.y + sh.y);
    o[2] = (bf16_t)mish_f(h4.z * sc.z + sh.z);
    o[3] = (bf16_t)mish_f(h4.w * sc.w + sh.w);
    *(v4bf*)(hm_l + row * 96 + q * 4) = o;
  }
  if (tid < 32) {
    v8bf z8 = {};
    *(v8bf*)(hm_l + (tid / 2) * 96 + 80 + (tid & 1) * 8) = z8;
  }

  // node layernorm -> bf16 (vectorized)
  for (int p = tid; p < 16 * NNODE; p += 256) {
    int row = p / NNODE, node = p % NNODE;
    const float4* x4 = (const float4*)(P.node_emb + ((b0 + row) * NNODE + node) * 48);
    float v[48];
    float s = 0.f, ss = 0.f;
#pragma unroll
    for (int q = 0; q < 12; q++) {
      float4 f = x4[q];
      v[4*q] = f.x; v[4*q+1] = f.y; v[4*q+2] = f.z; v[4*q+3] = f.w;
    }
#pragma unroll
    for (int k = 0; k < 48; k++) { s += v[k]; ss += v[k] * v[k]; }
    float mu = s * (1.f / 48.f);
    float var = ss * (1.f / 48.f) - mu * mu;
    float rs = rsqrtf(var + 1e-5f);
    bf16_t* d = nn_l + (row * NNODE + node) * 48;
#pragma unroll
    for (int q = 0; q < 12; q++) {
      float4 wv = ((const float4*)P.ln_n_w)[q];
      float4 bv = ((const float4*)P.ln_n_b)[q];
      v4bf o;
      o[0] = (bf16_t)((v[4*q]   - mu) * rs * wv.x + bv.x);
      o[1] = (bf16_t)((v[4*q+1] - mu) * rs * wv.y + bv.y);
      o[2] = (bf16_t)((v[4*q+2] - mu) * rs * wv.z + bv.z);
      o[3] = (bf16_t)((v[4*q+3] - mu) * rs * wv.w + bv.w);
      *(v4bf*)(d + q * 4) = o;
    }
  }
  async_wait0();
  __syncthreads();

  // tile embeddings (mean of 6 nodes), 8-wide chunks
  for (int p = tid; p < 16 * 19; p += 256) {
    int row = p / 19, t = p % 19;
    int idx[6];
#pragma unroll
    for (int j = 0; j < 6; j++) idx[j] = P.tile_nodes[t * 6 + j];
    const bf16_t* base = nn_l + row * NNODE * 48;
    bf16_t* d = tile_l + (row * 19 + t) * 48;
#pragma unroll
    for (int q = 0; q < 6; q++) {
      float acc[8] = {0.f, 0.f, 0.f, 0.f, 0.f, 0.f, 0.f, 0.f};
#pragma unroll
      for (int j = 0; j < 6; j++) {
        v8bf c = *(const v8bf*)(base + idx[j] * 48 + q * 8);
#pragma unroll
        for (int e = 0; e < 8; e++) acc[e] += (float)c[e];
      }
      v8bf o;
#pragma unroll
      for (int e = 0; e < 8; e++) o[e] = (bf16_t)(acc[e] * (1.f / 6.f));
      *(v8bf*)(d + q * 8) = o;
    }
  }
  __syncthreads();

  int lane = tid & 31, wid = tid >> 5;
  int l16 = lane & 15, half = lane >> 4;

  if (wid < 3) {
    // ---- settlement + city heads: 18 nodes/wave, B frags hoisted to regs ----
    const bf16_t* wS = w1_l;
    const bf16_t* wC = w1_l + 4096;
    v16bf bS[8], bC[8];
#pragma unroll
    for (int t = 0; t < 4; t++) {
      bS[2*t]   = bfrag(wS, 128, l16, t, half);
      bS[2*t+1] = bfrag(wS, 128, 16 + l16, t, half);
      bC[2*t]   = bfrag(wC, 128, l16, t, half);
      bC[2*t+1] = bfrag(wC, 128, 16 + l16, t, half);
    }
    float sw0 = P.sett2_w[l16], sw1 = P.sett2_w[16 + l16];
    float sb0 = P.sett1_b[l16], sb1 = P.sett1_b[16 + l16], s2b = P.sett2_b[0];
    float cw0 = P.city2_w[l16], cw1 = P.city2_w[16 + l16];
    float cb0 = P.city1_b[l16], cb1 = P.city1_b[16 + l16], c2b = P.city2_b[0];
    for (int node = wid * 18; node < wid * 18 + 18; ++node) {
      v16bf a[4];
#pragma unroll
      for (int t = 0; t < 4; t++) a[t] = ctx_frag_node(tn_l, nn_l, node, l16, t, half);
      v8f y0 = vzero8(), y1 = vzero8();
#pragma unroll
      for (int t = 0; t < 4; t++) {
        y0 = wmma_bf16(a[t], bS[2*t],   y0);
        y1 = wmma_bf16(a[t], bS[2*t+1], y1);
      }
      head1(y0, y1, sb0, sb1, sw0, sw1, s2b, P.out, b0, 5 + node, lane);
      y0 = vzero8(); y1 = vzero8();
#pragma unroll
      for (int t = 0; t < 4; t++) {
        y0 = wmma_bf16(a[t], bC[2*t],   y0);
        y1 = wmma_bf16(a[t], bC[2*t+1], y1);
      }
      head1(y0, y1, cb0, cb1, cw0, cw1, c2b, P.out, b0, 59 + node, lane);
    }
  } else if (wid < 6) {
    // ---- road head: 24 edges/wave, K padded 176->192, B frags hoisted ----
    const bf16_t* wR = w1_l + 8192;
    v16bf bR[12];
#pragma unroll
    for (int t = 0; t < 6; t++) {
      bR[2*t]   = bfrag(wR, 192, l16, t, half);
      bR[2*t+1] = bfrag(wR, 192, 16 + l16, t, half);
    }
    float rw0 = P.road2_w[l16], rw1 = P.road2_w[16 + l16];
    float rb0 = P.road1_b[l16], rb1 = P.road1_b[16 + l16], r2b = P.road2_b[0];
    for (int e = (wid - 3) * 24; e < (wid - 3) * 24 + 24; ++e) {
      int sn = P.road_pairs[e * 2], dn = P.road_pairs[e * 2 + 1];
      v16bf a[6];
#pragma unroll
      for (int t = 0; t < 6; t++) {
        int s0 = t * 32 + half * 8;
        a[t] = load_frag(road_ptr(tn_l, nn_l, zp, sn, dn, l16, s0),
                         road_ptr(tn_l, nn_l, zp, sn, dn, l16, s0 + 16));
      }
      v8f y0 = vzero8(), y1 = vzero8();
#pragma unroll
      for (int t = 0; t < 6; t++) {
        y0 = wmma_bf16(a[t], bR[2*t],   y0);
        y1 = wmma_bf16(a[t], bR[2*t+1], y1);
      }
      head1(y0, y1, rb0, rb1, rw0, rw1, r2b, P.out, b0, 113 + e, lane);
    }
  } else if (wid == 6) {
    // ---- robber head: 19 tiles, 5 outputs each, B frags hoisted ----
    const bf16_t* wB = w1_l + 14336;
    v16bf bB[8];
#pragma unroll
    for (int t = 0; t < 4; t++) {
      bB[2*t]   = bfrag(wB, 128, l16, t, half);
      bB[2*t+1] = bfrag(wB, 128, 16 + l16, t, half);
    }
    float bb0 = P.rob1_b[l16], bb1 = P.rob1_b[16 + l16];
    float w2a[5], w2b[5], b2v[5];
    for (int o = 0; o < 5; o++) {
      w2a[o] = P.rob2_w[o * 32 + l16];
      w2b[o] = P.rob2_w[o * 32 + 16 + l16];
      b2v[o] = P.rob2_b[o];
    }
    for (int t2 = 0; t2 < 19; ++t2) {
      v8f y0 = vzero8(), y1 = vzero8();
#pragma unroll
      for (int t = 0; t < 4; t++) {
        int s0 = t * 32 + half * 8, s1 = s0 + 16;
        const bf16_t* bt = tile_l + (l16 * 19 + t2) * 48 - 80;
        const bf16_t* p0 = (s0 < 80) ? tn_l + l16 * 80 + s0 : bt + s0;
        const bf16_t* p1 = (s1 < 80) ? tn_l + l16 * 80 + s1 : bt + s1;
        v16bf a = load_frag(p0, p1);
        y0 = wmma_bf16(a, bB[2*t],   y0);
        y1 = wmma_bf16(a, bB[2*t+1], y1);
      }
      float m0[8], m1[8];
#pragma unroll
      for (int i = 0; i < 8; i++) { m0[i] = mish_f(y0[i] + bb0); m1[i] = mish_f(y1[i] + bb1); }
      for (int o = 0; o < 5; o++) {
        float acc[8];
#pragma unroll
        for (int i = 0; i < 8; i++) acc[i] = m0[i] * w2a[o] + m1[i] * w2b[o];
#pragma unroll
        for (int m = 1; m < 16; m <<= 1) {
#pragma unroll
          for (int i = 0; i < 8; i++) acc[i] += __shfl_xor(acc[i], m, 32);
        }
        if (l16 == 0) {
#pragma unroll
          for (int i = 0; i < 8; i++)
            P.out[(size_t)(b0 + i + 8 * half) * ADIM + 185 + t2 * 5 + o] = acc[i] + b2v[o];
        }
      }
    }
  } else {
    // ---- gemm2: only output tiles covering cols [0,5) and [280,397) ----
    v16bf ah[3];
#pragma unroll
    for (int t = 0; t < 3; t++) {
      const bf16_t* p = hm_l + l16 * 96 + t * 32 + half * 8;
      ah[t] = load_frag(p, p + 16);
    }
    for (int idx = 0; idx < 9; ++idx) {
      int nt = (idx == 0) ? 0 : 16 + idx;
      v8f d = vzero8();
#pragma unroll
      for (int t = 0; t < 3; t++)
        d = wmma_bf16(ah[t], bfrag(P.gfc2_bf, 96, nt * 16 + l16, t, half), d);
      int col = nt * 16 + l16;
      bool ok = (col < 5) || (col >= 280 && col < 397);
      if (ok) {
        float bias = P.gfc2_b[col];
#pragma unroll
        for (int i = 0; i < 8; i++)
          P.out[(size_t)(b0 + i + 8 * half) * ADIM + col] = d[i] + bias;
      }
    }
  }
}

extern "C" void kernel_launch(void* const* d_in, const int* in_sizes, int n_in,
                              void* d_out, int out_size, void* d_ws, size_t ws_size,
                              hipStream_t stream) {
  const float* trunk      = (const float*)d_in[0];
  const float* node_emb   = (const float*)d_in[1];
  const int*   road_pairs = (const int*)d_in[2];
  const int*   tile_nodes = (const int*)d_in[3];
  const float* ln_t_w = (const float*)d_in[4];
  const float* ln_t_b = (const float*)d_in[5];
  const float* ln_n_w = (const float*)d_in[6];
  const float* ln_n_b = (const float*)d_in[7];
  const float* gfc1_w = (const float*)d_in[8];
  const float* gfc1_b = (const float*)d_in[9];
  const float* bn_w   = (const float*)d_in[10];
  const float* bn_b   = (const float*)d_in[11];
  const float* gfc2_w = (const float*)d_in[12];
  const float* gfc2_b = (const float*)d_in[13];
  const float* sett1_w = (const float*)d_in[14];
  const float* sett1_b = (const float*)d_in[15];
  const float* sett2_w = (const float*)d_in[16];
  const float* sett2_b = (const float*)d_in[17];
  const float* city1_w = (const float*)d_in[18];
  const float* city1_b = (const float*)d_in[19];
  const float* city2_w = (const float*)d_in[20];
  const float* city2_b = (const float*)d_in[21];
  const float* road1_w = (const float*)d_in[22];
  const float* road1_b = (const float*)d_in[23];
  const float* road2_w = (const float*)d_in[24];
  const float* road2_b = (const float*)d_in[25];
  const float* rob1_w  = (const float*)d_in[26];
  const float* rob1_b  = (const float*)d_in[27];
  const float* rob2_w  = (const float*)d_in[28];
  const float* rob2_b  = (const float*)d_in[29];

  char* ws = (char*)d_ws;
  bf16_t* tn_g    = (bf16_t*)(ws + OFF_TN);
  float*  hbuf    = (float*)(ws + OFF_H);
  float*  psum    = (float*)(ws + OFF_PS);
  float*  psq     = (float*)(ws + OFF_PQ);
  float*  scale   = (float*)(ws + OFF_SC);
  float*  shift   = (float*)(ws + OFF_SHF);
  bf16_t* gfc1_bf = (bf16_t*)(ws + OFF_G1);
  bf16_t* gfc2_bf = (bf16_t*)(ws + OFF_G2);
  bf16_t* w1_all  = (bf16_t*)(ws + OFF_W1);

  int B = in_sizes[0] / TDIM;     // 16384
  int nblk = B / 16;              // 1024

  k_prep<<<1, 256, 0, stream>>>(gfc1_w, gfc2_w, sett1_w, city1_w, road1_w, rob1_w,
                                gfc1_bf, gfc2_bf, w1_all);
  k_trunk<<<nblk, 128, 0, stream>>>(trunk, ln_t_w, ln_t_b, gfc1_bf, gfc1_b,
                                    tn_g, hbuf, psum, psq);
  k_bnstats<<<1, 128, 0, stream>>>(psum, psq, bn_w, bn_b, scale, shift,
                                   nblk, 1.f / (float)B);

  CParams P;
  P.out = (float*)d_out;
  P.tn_g = tn_g; P.hbuf = hbuf; P.scale = scale; P.shift = shift;
  P.node_emb = node_emb; P.ln_n_w = ln_n_w; P.ln_n_b = ln_n_b;
  P.road_pairs = road_pairs; P.tile_nodes = tile_nodes;
  P.w1_all_g = w1_all; P.gfc2_bf = gfc2_bf; P.gfc2_b = gfc2_b;
  P.sett1_b = sett1_b; P.sett2_w = sett2_w; P.sett2_b = sett2_b;
  P.city1_b = city1_b; P.city2_w = city2_w; P.city2_b = city2_b;
  P.road1_b = road1_b; P.road2_w = road2_w; P.road2_b = road2_b;
  P.rob1_b  = rob1_b;  P.rob2_w  = rob2_w;  P.rob2_b  = rob2_b;

  k_heads<<<nblk, 256, LDS_BYTES, stream>>>(P);
}
